// MultiHeadAttention_65773129171319
// MI455X (gfx1250) — matrix operand order repaired and checked by reading, not compiled, not run on hardware
//
#include <hip/hip_runtime.h>
#include <hip/hip_bf16.h>
#include <math.h>

typedef __bf16 bf16_t;
typedef __attribute__((ext_vector_type(16))) __bf16 v16bf;
typedef __attribute__((ext_vector_type(8)))  __bf16 v8bf;
typedef __attribute__((ext_vector_type(8)))  float   v8f;

#define B_   2
#define S_   2048
#define D_   512
#define H_   8
#define DK_  64
#define ROWS (B_*S_)              // 4096
#define INV_SCALE 0.0883883476483184f   // 1/sqrt(2*DK) = 1/sqrt(128)

// ---------------------------------------------------------------------------
// WMMA helper: D = A(16x32 bf16) * B(32x16 bf16) + C(16x16 f32)
// ---------------------------------------------------------------------------
__device__ __forceinline__ v8f wmma_bf16(v16bf a, v16bf b, v8f c) {
  return __builtin_amdgcn_wmma_f32_16x16x32_bf16(
      /*neg_a=*/false, a, /*neg_b=*/false, b,
      /*c_mod=*/(short)0, c, /*reuse_a=*/false, /*reuse_b=*/false);
}

__device__ __forceinline__ v16bf combine16(v8bf lo, v8bf hi) {
  v16bf r;
#pragma unroll
  for (int i = 0; i < 8; ++i) { r[i] = lo[i]; r[i + 8] = hi[i]; }
  return r;
}

// ---------------------------------------------------------------------------
// Kernel 0: weight f32 [k][n] -> bf16 transposed [n][k]
// ---------------------------------------------------------------------------
__global__ void wconv_kernel(const float* __restrict__ w, bf16_t* __restrict__ wt) {
  int id = blockIdx.x * blockDim.x + threadIdx.x;   // 0 .. 512*512-1
  int n = id >> 9;
  int k = id & 511;
  wt[(size_t)n * 512 + k] = (bf16_t)w[(size_t)k * 512 + n];
}

// ---------------------------------------------------------------------------
// Kernel 1/3: WMMA GEMM  Y[4096x512] = A[4096x512] @ W (W given transposed bf16)
// Each wave computes a 16-row x 64-col tile. 2048 waves = 256 blocks x 8 waves.
// A_F32:   A is f32 (convert to bf16 on load) else A is bf16.
// OUT_MODE 0: bf16 [row][col]   (projection outputs q/k)
//          1: bf16 transposed per head [b,h,dk,s]  (v outputs)
//          2: f32  [row][col]   (final outputs into d_out)
// ---------------------------------------------------------------------------
template <bool A_F32, int OUT_MODE>
__global__ void __launch_bounds__(256)
gemm512_kernel(const void* __restrict__ Ap, const bf16_t* __restrict__ WT,
               void* __restrict__ outp) {
  const int wave = (int)((blockIdx.x * blockDim.x + threadIdx.x) >> 5);
  const int lane = threadIdx.x & 31;
  const int row_tile = wave >> 3;   // 0..255  (16 rows each)
  const int cg       = wave & 7;    // column group of 64
  const int hi16 = lane >> 4;
  const int n    = lane & 15;
  const int kb0  = hi16 * 8;        // A-frag K base for this lane half
  const int arow = row_tile * 16 + n;

  v8f acc[4];
#pragma unroll
  for (int t = 0; t < 4; ++t)
#pragma unroll
    for (int i = 0; i < 8; ++i) acc[t][i] = 0.0f;

#pragma unroll 4
  for (int step = 0; step < 16; ++step) {
    v16bf afrag;
    if (A_F32) {
      const float* ab = (const float*)Ap + (size_t)arow * D_ + step * 32 + kb0;
      float4 f0 = *(const float4*)(ab);
      float4 f1 = *(const float4*)(ab + 4);
      float4 f2 = *(const float4*)(ab + 16);
      float4 f3 = *(const float4*)(ab + 20);
      afrag[0]  = (bf16_t)f0.x; afrag[1]  = (bf16_t)f0.y;
      afrag[2]  = (bf16_t)f0.z; afrag[3]  = (bf16_t)f0.w;
      afrag[4]  = (bf16_t)f1.x; afrag[5]  = (bf16_t)f1.y;
      afrag[6]  = (bf16_t)f1.z; afrag[7]  = (bf16_t)f1.w;
      afrag[8]  = (bf16_t)f2.x; afrag[9]  = (bf16_t)f2.y;
      afrag[10] = (bf16_t)f2.z; afrag[11] = (bf16_t)f2.w;
      afrag[12] = (bf16_t)f3.x; afrag[13] = (bf16_t)f3.y;
      afrag[14] = (bf16_t)f3.z; afrag[15] = (bf16_t)f3.w;
    } else {
      const bf16_t* ab = (const bf16_t*)Ap + (size_t)arow * D_ + step * 32 + kb0;
      afrag = combine16(*(const v8bf*)ab, *(const v8bf*)(ab + 16));
    }
#pragma unroll
    for (int t = 0; t < 4; ++t) {
      const bf16_t* wb = WT + (size_t)(cg * 64 + t * 16 + n) * D_ + step * 32 + hi16 * 8;
      v16bf bfrag = combine16(*(const v8bf*)wb, *(const v8bf*)(wb + 16));
      acc[t] = wmma_bf16(afrag, bfrag, acc[t]);
    }
  }

#pragma unroll
  for (int t = 0; t < 4; ++t)
#pragma unroll
    for (int r = 0; r < 8; ++r) {
      const int row = row_tile * 16 + r + 8 * hi16;
      const int col = cg * 64 + t * 16 + n;
      const float v = acc[t][r];
      if (OUT_MODE == 0) {
        ((bf16_t*)outp)[(size_t)row * D_ + col] = (bf16_t)v;
      } else if (OUT_MODE == 2) {
        ((float*)outp)[(size_t)row * D_ + col] = v;
      } else {  // transposed per-head: [b, h=cg, dk, s]
        const int bb = row >> 11;           // row / S_
        const int s  = row & (S_ - 1);
        const int dk = t * 16 + n;
        ((bf16_t*)outp)[(((size_t)(bb * H_ + cg)) * DK_ + dk) * S_ + s] = (bf16_t)v;
      }
    }
}

// ---------------------------------------------------------------------------
// Kernel 2: fused complex attention, flash-style. 1 wave = 16 q-rows of (b,h).
// 2048 waves = 256 blocks x 8 waves. Per-wave LDS P-buffer for C->A relayout.
// ---------------------------------------------------------------------------
__global__ void __launch_bounds__(256)
attn_kernel(const bf16_t* __restrict__ qr, const bf16_t* __restrict__ qp,
            const bf16_t* __restrict__ kr, const bf16_t* __restrict__ kp,
            const bf16_t* __restrict__ vrT, const bf16_t* __restrict__ vpT,
            const int* __restrict__ mask,
            bf16_t* __restrict__ xr, bf16_t* __restrict__ xp) {
  __shared__ alignas(16) bf16_t Pbuf[8][16 * 32];

  const int wave   = (int)((blockIdx.x * blockDim.x + threadIdx.x) >> 5);
  const int wlocal = threadIdx.x >> 5;
  const int lane   = threadIdx.x & 31;
  const int qt = wave & 127;          // S_/16 = 128 q-tiles
  const int h  = (wave >> 7) & (H_ - 1);
  const int b  = wave >> 10;
  const int qb = qt * 16;
  const int hi16 = lane >> 4;
  const int n    = lane & 15;
  const int kb0  = hi16 * 8;
  bf16_t* P = &Pbuf[wlocal][0];

  const size_t qkoff = (size_t)b * S_ * D_ + (size_t)h * DK_;
  const bf16_t* qrb = qr + qkoff;
  const bf16_t* qpb = qp + qkoff;
  const bf16_t* krb = kr + qkoff;
  const bf16_t* kpb = kp + qkoff;
  const size_t voff = (size_t)(b * H_ + h) * DK_ * S_;
  const bf16_t* vrb = vrT + voff;
  const bf16_t* vpb = vpT + voff;
  const int* mb = mask + (size_t)b * S_ * S_;

  // --- load Q fragments (persist in registers) ---
  v16bf Qr[2], Qp[2];
  const int qrow = qb + n;
#pragma unroll
  for (int st = 0; st < 2; ++st) {
    const bf16_t* p1 = qrb + (size_t)qrow * D_ + st * 32 + kb0;
    Qr[st] = combine16(*(const v8bf*)p1, *(const v8bf*)(p1 + 16));
    const bf16_t* p2 = qpb + (size_t)qrow * D_ + st * 32 + kb0;
    Qp[st] = combine16(*(const v8bf*)p2, *(const v8bf*)(p2 + 16));
  }

  v8f Or[4], Op[4];
#pragma unroll
  for (int t = 0; t < 4; ++t)
#pragma unroll
    for (int i = 0; i < 8; ++i) { Or[t][i] = 0.0f; Op[t][i] = 0.0f; }
  float m[8], l[8];
#pragma unroll
  for (int r = 0; r < 8; ++r) { m[r] = -1e30f; l[r] = 0.0f; }

  for (int kb = 0; kb < S_; kb += 32) {
    float sc[2][8];
#pragma unroll
    for (int kt = 0; kt < 2; ++kt) {
      v8f aR, aT, aP;
#pragma unroll
      for (int i = 0; i < 8; ++i) { aR[i] = 0.0f; aT[i] = 0.0f; aP[i] = 0.0f; }
#pragma unroll
      for (int st = 0; st < 2; ++st) {
        const bf16_t* krp = krb + (size_t)(kb + kt * 16 + n) * D_ + st * 32 + hi16 * 8;
        v16bf Bkr = combine16(*(const v8bf*)krp, *(const v8bf*)(krp + 16));
        const bf16_t* kpp = kpb + (size_t)(kb + kt * 16 + n) * D_ + st * 32 + hi16 * 8;
        v16bf Bkp = combine16(*(const v8bf*)kpp, *(const v8bf*)(kpp + 16));
        aR = wmma_bf16(Qr[st], Bkr, aR);   // qr . kr
        aT = wmma_bf16(Qp[st], Bkp, aT);   // qp . kp
        aP = wmma_bf16(Qr[st], Bkp, aP);   // qr . kp
        aP = wmma_bf16(Qp[st], Bkr, aP);   //  + qp . kr
      }
      const int kcol = kb + kt * 16 + n;
#pragma unroll
      for (int r = 0; r < 8; ++r) {
        const float srl = aR[r] - aT[r];
        const float spl = aP[r];
        const float v = sqrtf(srl * srl + spl * spl) * INV_SCALE;
        const int q = qb + r + 8 * hi16;
        const int mk = mb[(size_t)q * S_ + kcol];
        sc[kt][r] = (mk == 0) ? -1e9f : v;
      }
    }

    // --- online softmax update (row stats across 16-lane half-groups) ---
#pragma unroll
    for (int r = 0; r < 8; ++r) {
      float rmax = fmaxf(sc[0][r], sc[1][r]);
#pragma unroll
      for (int o = 1; o < 16; o <<= 1) rmax = fmaxf(rmax, __shfl_xor(rmax, o, 32));
      const float mn   = fmaxf(m[r], rmax);
      const float p0   = __expf(sc[0][r] - mn);
      const float p1   = __expf(sc[1][r] - mn);
      const float corr = __expf(m[r] - mn);
      float ps = p0 + p1;
#pragma unroll
      for (int o = 1; o < 16; o <<= 1) ps += __shfl_xor(ps, o, 32);
      l[r] = l[r] * corr + ps;
      m[r] = mn;
#pragma unroll
      for (int t = 0; t < 4; ++t) { Or[t][r] *= corr; Op[t][r] *= corr; }
      const int row = r + 8 * hi16;
      P[row * 32 + n]      = (bf16_t)p0;
      P[row * 32 + 16 + n] = (bf16_t)p1;
    }
    // wave-local LDS RAW: wait for our ds stores before cross-lane reload
    asm volatile("s_wait_dscnt 0x0" ::: "memory");

    // reload P as a 16x32 A fragment
    const bf16_t* pp = P + (size_t)n * 32 + kb0;
    v16bf Pa = combine16(*(const v8bf*)pp, *(const v8bf*)(pp + 16));

    // O += P @ V  (V pre-transposed: contiguous B-fragment loads)
#pragma unroll
    for (int t = 0; t < 4; ++t) {
      const bf16_t* vrp = vrb + (size_t)(t * 16 + n) * S_ + kb + hi16 * 8;
      v16bf Bvr = combine16(*(const v8bf*)vrp, *(const v8bf*)(vrp + 16));
      const bf16_t* vpp = vpb + (size_t)(t * 16 + n) * S_ + kb + hi16 * 8;
      v16bf Bvp = combine16(*(const v8bf*)vpp, *(const v8bf*)(vpp + 16));
      Or[t] = wmma_bf16(Pa, Bvr, Or[t]);
      Op[t] = wmma_bf16(Pa, Bvp, Op[t]);
    }
  }

  // --- epilogue: normalize and store bf16 [B,S,D] ---
#pragma unroll
  for (int r = 0; r < 8; ++r) {
    const float inv = 1.0f / l[r];
    const int q = qb + r + 8 * hi16;
    const size_t base = ((size_t)b * S_ + q) * D_ + (size_t)h * DK_;
#pragma unroll
    for (int t = 0; t < 4; ++t) {
      xr[base + t * 16 + n] = (bf16_t)(Or[t][r] * inv);
      xp[base + t * 16 + n] = (bf16_t)(Op[t][r] * inv);
    }
  }
}

// ---------------------------------------------------------------------------
// Host launcher
// ---------------------------------------------------------------------------
extern "C" void kernel_launch(void* const* d_in, const int* in_sizes, int n_in,
                              void* d_out, int out_size, void* d_ws, size_t ws_size,
                              hipStream_t stream) {
  (void)in_sizes; (void)n_in; (void)out_size; (void)ws_size;
  const float* q_real  = (const float*)d_in[0];
  const float* k_real  = (const float*)d_in[1];
  const float* v_real  = (const float*)d_in[2];
  const float* q_phase = (const float*)d_in[3];
  const float* k_phase = (const float*)d_in[4];
  const float* v_phase = (const float*)d_in[5];
  const float* w_q     = (const float*)d_in[6];
  const float* w_k     = (const float*)d_in[7];
  const float* w_v     = (const float*)d_in[8];
  const float* w_o     = (const float*)d_in[9];
  const int*   maskp   = (const int*)d_in[10];

  char* ws = (char*)d_ws;
  const size_t WSZ = (size_t)512 * 512 * sizeof(bf16_t);   // 512 KB
  const size_t TSZ = (size_t)ROWS * D_ * sizeof(bf16_t);   // 4 MB
  bf16_t* wqT = (bf16_t*)(ws + 0 * WSZ);
  bf16_t* wkT = (bf16_t*)(ws + 1 * WSZ);
  bf16_t* wvT = (bf16_t*)(ws + 2 * WSZ);
  bf16_t* woT = (bf16_t*)(ws + 3 * WSZ);
  char* tb = ws + 4 * WSZ;
  bf16_t* qrB  = (bf16_t*)(tb + 0 * TSZ);
  bf16_t* qpB  = (bf16_t*)(tb + 1 * TSZ);
  bf16_t* krB  = (bf16_t*)(tb + 2 * TSZ);
  bf16_t* kpB  = (bf16_t*)(tb + 3 * TSZ);
  bf16_t* vrTB = (bf16_t*)(tb + 4 * TSZ);
  bf16_t* vpTB = (bf16_t*)(tb + 5 * TSZ);
  bf16_t* xrB  = (bf16_t*)(tb + 6 * TSZ);
  bf16_t* xpB  = (bf16_t*)(tb + 7 * TSZ);

  // 0) weights -> transposed bf16
  wconv_kernel<<<1024, 256, 0, stream>>>(w_q, wqT);
  wconv_kernel<<<1024, 256, 0, stream>>>(w_k, wkT);
  wconv_kernel<<<1024, 256, 0, stream>>>(w_v, wvT);
  wconv_kernel<<<1024, 256, 0, stream>>>(w_o, woT);

  // 1) projections (2048 waves -> 256 blocks x 256 threads)
  gemm512_kernel<true, 0><<<256, 256, 0, stream>>>(q_real,  wqT, qrB);
  gemm512_kernel<true, 0><<<256, 256, 0, stream>>>(q_phase, wqT, qpB);
  gemm512_kernel<true, 0><<<256, 256, 0, stream>>>(k_real,  wkT, krB);
  gemm512_kernel<true, 0><<<256, 256, 0, stream>>>(k_phase, wkT, kpB);
  gemm512_kernel<true, 1><<<256, 256, 0, stream>>>(v_real,  wvT, vrTB);
  gemm512_kernel<true, 1><<<256, 256, 0, stream>>>(v_phase, wvT, vpTB);

  // 2) fused complex attention (2048 waves)
  attn_kernel<<<256, 256, 0, stream>>>(qrB, qpB, krB, kpB, vrTB, vpTB, maskp, xrB, xpB);

  // 3) output projections -> f32 d_out (xr result first, xp second)
  float* out0 = (float*)d_out;
  float* out1 = out0 + (size_t)ROWS * D_;
  gemm512_kernel<false, 2><<<256, 256, 0, stream>>>(xrB, woT, out0);
  gemm512_kernel<false, 2><<<256, 256, 0, stream>>>(xpB, woT, out1);
}
